// Head_32100585570892
// MI455X (gfx1250) — matrix operand order repaired and checked by reading, not compiled, not run on hardware
//
#include <hip/hip_runtime.h>

#define B_ 8
#define T_ 2048
#define E_ 768
#define H_ 64

typedef _Float16 half_t;
typedef __attribute__((ext_vector_type(16))) _Float16 v16h;
typedef __attribute__((ext_vector_type(8)))  _Float16 v8h;
typedef __attribute__((ext_vector_type(4)))  _Float16 v4h;
typedef __attribute__((ext_vector_type(8)))  float    v8f;
typedef __attribute__((ext_vector_type(4)))  float    v4f;

union AFrag { v16h v; v8h h[2]; _Float16 e[16]; };

// A fragment (16xK, K-slice of 32): lane m=lane&15, g=lane>>4,
// elements 0..15 -> K = kbase+16g .. +15 (contiguous, matches B fragment)
__device__ inline v16h load_a_frag(const half_t* base, int ld, int kbase, int lane) {
  int m = lane & 15, g = lane >> 4;
  AFrag f;
  const half_t* p = base + m * ld + kbase;
  f.h[0] = *(const v8h*)(p + 16 * g);
  f.h[1] = *(const v8h*)(p + 16 * g + 8);
  return f.v;
}

// B fragment (32x16): lane n=lane&15 owns memory row (row_off+n); per-lane
// 16 contiguous elements at k = kbase + e + 16g (contiguous along k in memory)
__device__ inline v16h load_b_frag(const half_t* base, int ld, int row_off, int kbase, int lane) {
  int n = lane & 15, g = lane >> 4;
  AFrag f;
  const half_t* p = base + (size_t)(row_off + n) * ld + kbase + 16 * g;
  f.h[0] = *(const v8h*)(p);
  f.h[1] = *(const v8h*)(p + 8);
  return f.v;
}

__device__ inline v8f wmma_f16(v16h a, v16h b, v8f c) {
  return __builtin_amdgcn_wmma_f32_16x16x32_f16(false, a, false, b, (short)0, c, false, false);
}

// ------------- Phase 0: one-time W f32 -> f16 transpose ([E][H] -> [H][E]) ---
__global__ void __launch_bounds__(256) wcvt_kernel(
    const float* __restrict__ Wq, const float* __restrict__ Wk,
    const float* __restrict__ Wv, half_t* __restrict__ Wt) {
  int i = blockIdx.x * 256 + threadIdx.x;     // over E_*H_
  if (i >= E_ * H_) return;
  int k = i / H_, h = i % H_;
  Wt[(size_t)h * E_ + k]            = (_Float16)Wq[i];
  Wt[(size_t)(H_ + h) * E_ + k]     = (_Float16)Wk[i];
  Wt[(size_t)(2 * H_ + h) * E_ + k] = (_Float16)Wv[i];
}

// ---------------- Phase 1: QKV projection with WMMA, V stored transposed -----
__global__ void __launch_bounds__(128) qkv_proj_kernel(
    const float* __restrict__ x, const half_t* __restrict__ Wt,
    half_t* __restrict__ Qw, half_t* __restrict__ Kw, half_t* __restrict__ Vt) {
  __shared__ alignas(16) half_t xt[16 * E_];
  int blk = blockIdx.x;
  int b  = blk >> 7;          // T/16 = 128 tiles per batch
  int t0 = (blk & 127) << 4;
  int tid = threadIdx.x;

  // cooperative load of x tile [16 x 768] f32 -> f16 LDS
  const float* xp = x + (size_t)(b * T_ + t0) * E_;
#pragma unroll
  for (int i = 0; i < 24; ++i) {
    int idx = tid + i * 128;      // 0..3071 float4s
    int row = idx / 192;          // 768/4 = 192
    int c4  = idx % 192;
    v4f v = *(const v4f*)(xp + row * E_ + c4 * 4);
    v4h h;
    h[0] = (_Float16)v[0]; h[1] = (_Float16)v[1];
    h[2] = (_Float16)v[2]; h[3] = (_Float16)v[3];
    *(v4h*)(xt + row * E_ + c4 * 4) = h;
  }
  __syncthreads();

  int lane  = tid & 31;
  int ntile = tid >> 5;                 // 4 waves -> 4 slices of H
  int g = lane >> 4, n = lane & 15;
  int col = ntile * 16 + n;

  const half_t* Wtq = Wt;
  const half_t* Wtk = Wt + (size_t)H_ * E_;
  const half_t* Wtv = Wt + (size_t)2 * H_ * E_;

  v8f cq = {}, ck = {}, cv = {};
  for (int kk = 0; kk < E_ / 32; ++kk) {
    int kbase = kk * 32;
    v16h a  = load_a_frag(xt, E_, kbase, lane);
    v16h bq = load_b_frag(Wtq, E_, ntile * 16, kbase, lane);
    cq = wmma_f16(a, bq, cq);
    v16h bk = load_b_frag(Wtk, E_, ntile * 16, kbase, lane);
    ck = wmma_f16(a, bk, ck);
    v16h bv = load_b_frag(Wtv, E_, ntile * 16, kbase, lane);
    cv = wmma_f16(a, bv, cv);
  }

#pragma unroll
  for (int r = 0; r < 8; ++r) {
    int t = t0 + r + 8 * g;
    Qw[(size_t)(b * T_ + t) * H_ + col] = (_Float16)(cq[r] * 0.125f);  // bake H^-0.5
    Kw[(size_t)(b * T_ + t) * H_ + col] = (_Float16)ck[r];
    Vt[((size_t)b * H_ + col) * T_ + t] = (_Float16)cv[r];             // transposed
  }
}

// ---------------- Phase 2: flash attention, one wave per 16-query tile -------
__global__ void __launch_bounds__(32) attn_kernel(
    const half_t* __restrict__ Qw, const half_t* __restrict__ Kw,
    const half_t* __restrict__ Vt, float* __restrict__ out) {
  __shared__ alignas(16) half_t Pl[16 * 32];
  int blk = blockIdx.x;
  int b  = blk >> 7;
  int qt = blk & 127;
  int q0 = qt << 4;
  int lane = threadIdx.x & 31;
  int g = lane >> 4, n = lane & 15;

  const half_t* Qb = Qw + (size_t)b * T_ * H_;
  const half_t* Kb = Kw + (size_t)b * T_ * H_;
  const half_t* Vb = Vt + (size_t)b * H_ * T_;

  // Q fragments for the whole tile, kept in registers (K-dim 64 = two slices)
  v16h aQ0 = load_a_frag(Qb + (size_t)q0 * H_, H_, 0,  lane);
  v16h aQ1 = load_a_frag(Qb + (size_t)q0 * H_, H_, 32, lane);

  v8f O0 = {}, O1 = {}, O2 = {}, O3 = {};
  float mrow[8], lrow[8];
#pragma unroll
  for (int r = 0; r < 8; ++r) { mrow[r] = -1e30f; lrow[r] = 0.0f; }

  int nblk = ((q0 + 15) >> 5) + 1;   // 32-key blocks needed under causality
  for (int kb = 0; kb < nblk; ++kb) {
    int k0 = kb << 5;

    // prefetch next key block's K rows and V^T rows into caches (gfx1250)
    if (kb + 1 < nblk) {
      int kn = k0 + 32;
      __builtin_prefetch(Kb + (size_t)(kn + lane) * H_, 0, 0);       // 32 K rows
      __builtin_prefetch(Vb + (size_t)lane * T_ + kn, 0, 0);         // V^T rows 0..31
      __builtin_prefetch(Vb + (size_t)(32 + lane) * T_ + kn, 0, 0);  // V^T rows 32..63
    }

    // S = Q * K^T : two 16-key tiles, K-dim 64 chained in two WMMAs each
    v16h bK00 = load_b_frag(Kb, H_, k0,      0,  lane);
    v16h bK01 = load_b_frag(Kb, H_, k0,      32, lane);
    v8f S0 = {};
    S0 = wmma_f16(aQ0, bK00, S0);
    S0 = wmma_f16(aQ1, bK01, S0);
    v16h bK10 = load_b_frag(Kb, H_, k0 + 16, 0,  lane);
    v16h bK11 = load_b_frag(Kb, H_, k0 + 16, 32, lane);
    v8f S1 = {};
    S1 = wmma_f16(aQ0, bK10, S1);
    S1 = wmma_f16(aQ1, bK11, S1);

    bool diag = (k0 + 31 > q0);      // only diagonal blocks need masking
    int s0c = k0 + n;
    int s1c = k0 + 16 + n;

#pragma unroll
    for (int r = 0; r < 8; ++r) {
      int t = q0 + r + 8 * g;
      float v0 = S0[r], v1 = S1[r];
      if (diag) {                    // wave-uniform branch; select, no EXEC change
        v0 = (s0c <= t) ? v0 : -1e30f;
        v1 = (s1c <= t) ? v1 : -1e30f;
      }
      // row max across the 16 lanes of this half-wave (row lives in one half)
      float mx = fmaxf(v0, v1);
      mx = fmaxf(mx, __shfl_xor(mx, 1, 32));
      mx = fmaxf(mx, __shfl_xor(mx, 2, 32));
      mx = fmaxf(mx, __shfl_xor(mx, 4, 32));
      mx = fmaxf(mx, __shfl_xor(mx, 8, 32));
      float mnew  = fmaxf(mrow[r], mx);
      float alpha = __expf(mrow[r] - mnew);
      float p0 = __expf(v0 - mnew);
      float p1 = __expf(v1 - mnew);
      float s = p0 + p1;
      s += __shfl_xor(s, 1, 32);
      s += __shfl_xor(s, 2, 32);
      s += __shfl_xor(s, 4, 32);
      s += __shfl_xor(s, 8, 32);
      lrow[r] = lrow[r] * alpha + s;
      mrow[r] = mnew;
      O0[r] *= alpha; O1[r] *= alpha; O2[r] *= alpha; O3[r] *= alpha;
      int m = r + 8 * g;
      Pl[m * 32 + n]      = (_Float16)p0;   // C-layout -> LDS (A-layout reload)
      Pl[m * 32 + 16 + n] = (_Float16)p1;
    }
    asm volatile("s_wait_dscnt 0" ::: "memory");  // LDS RAW across lanes, same wave

    v16h aP = load_a_frag(Pl, 32, 0, lane);       // P as 16x32 A fragment
    // O += P * V  (V transposed -> contiguous B fragments)
    v16h bV0 = load_b_frag(Vb, T_, 0,  k0, lane);
    O0 = wmma_f16(aP, bV0, O0);
    v16h bV1 = load_b_frag(Vb, T_, 16, k0, lane);
    O1 = wmma_f16(aP, bV1, O1);
    v16h bV2 = load_b_frag(Vb, T_, 32, k0, lane);
    O2 = wmma_f16(aP, bV2, O2);
    v16h bV3 = load_b_frag(Vb, T_, 48, k0, lane);
    O3 = wmma_f16(aP, bV3, O3);
  }

  float* ob = out + (size_t)b * T_ * H_;
#pragma unroll
  for (int r = 0; r < 8; ++r) {
    float inv = 1.0f / lrow[r];
    int t = q0 + r + 8 * g;
    ob[(size_t)t * H_ + n]      = O0[r] * inv;
    ob[(size_t)t * H_ + 16 + n] = O1[r] * inv;
    ob[(size_t)t * H_ + 32 + n] = O2[r] * inv;
    ob[(size_t)t * H_ + 48 + n] = O3[r] * inv;
  }
}

extern "C" void kernel_launch(void* const* d_in, const int* in_sizes, int n_in,
                              void* d_out, int out_size, void* d_ws, size_t ws_size,
                              hipStream_t stream) {
  const float* x  = (const float*)d_in[0];
  const float* Wq = (const float*)d_in[1];
  const float* Wk = (const float*)d_in[2];
  const float* Wv = (const float*)d_in[3];

  half_t* Qw = (half_t*)d_ws;                        // [B][T][64] f16, pre-scaled
  half_t* Kw = Qw + (size_t)B_ * T_ * H_;            // [B][T][64] f16
  half_t* Vt = Kw + (size_t)B_ * T_ * H_;            // [B][64][T] f16 (transposed)
  half_t* Wt = Vt + (size_t)B_ * T_ * H_;            // [3][64][768] f16 (transposed)

  wcvt_kernel<<<(E_ * H_ + 255) / 256, 256, 0, stream>>>(Wq, Wk, Wv, Wt);

  dim3 grid(B_ * (T_ / 16));                         // 1024 tiles
  qkv_proj_kernel<<<grid, 128, 0, stream>>>(x, Wt, Qw, Kw, Vt);
  attn_kernel<<<grid, 32, 0, stream>>>(Qw, Kw, Vt, (float*)d_out);
}
